// AnchoredSelfAttention_43911745635107
// MI455X (gfx1250) — matrix operand order repaired and checked by reading, not compiled, not run on hardware
//
#include <hip/hip_runtime.h>
#include <hip/hip_bf16.h>

// Anchored self-attention, MI455X (gfx1250, wave32, WMMA + async LDS loads).
// B=8, C=256, H=W=64, N=4096.
// A-layout: lane = M-row, two 8-element groups at K = koff and koff+16.
// B-layout: lane = N-col, two 8-element groups at K = koff and koff+16.
// Staged LDS tiles are N-major with K contiguous -> all fragments are b128 pairs.
// Global->LDS staging uses GLOBAL_LOAD_ASYNC_TO_LDS_B128 (ASYNCcnt) with
// double-buffered K-chunks so staging overlaps the WMMA pipeline.

#define BB 8
#define CC 256
#define NN 4096

typedef __attribute__((ext_vector_type(16))) __bf16 v16bf;
typedef __attribute__((ext_vector_type(8)))  float  v8f;
typedef __attribute__((ext_vector_type(4)))  unsigned int v4u;
typedef __attribute__((ext_vector_type(4)))  int v4i;

#define AS1 __attribute__((address_space(1)))
#define AS3 __attribute__((address_space(3)))

union ABu { v16bf v; __bf16 h[16]; v4u q[2]; };

// One 16-byte async copy, global -> LDS (per-lane), tracked by ASYNCcnt.
// Builtin signature (from clang diagnostic): (v4i AS1*, v4i AS3*, imm offset, imm cpol).
__device__ __forceinline__ void async_cp_b128(__bf16* dst, const __bf16* src){
#if __has_builtin(__builtin_amdgcn_global_load_async_to_lds_b128)
  __bf16* s0      = (__bf16*)src;          // drop const
  AS1 __bf16* s1  = (AS1 __bf16*)s0;       // generic -> global AS
  AS3 __bf16* d3  = (AS3 __bf16*)dst;      // generic -> LDS AS
  __builtin_amdgcn_global_load_async_to_lds_b128(
      (AS1 v4i*)s1, (AS3 v4i*)d3, 0, 0);
#else
  asm volatile("global_load_async_to_lds_b128 %0, %1, off"
               :: "v"((unsigned)(size_t)(AS3 __bf16*)dst),
                  "v"((unsigned long long)src)
               : "memory");
#endif
}

__device__ __forceinline__ void wait_async0(){
#if __has_builtin(__builtin_amdgcn_s_wait_asynccnt)
  __builtin_amdgcn_s_wait_asynccnt(0);
#else
  asm volatile("s_wait_asynccnt 0x0" ::: "memory");
#endif
}

__device__ __forceinline__ float half_max16(float v){
#pragma unroll
  for (int m = 1; m < 16; m <<= 1) v = fmaxf(v, __shfl_xor(v, m, 32));
  return v;
}
__device__ __forceinline__ float half_sum16(float v){
#pragma unroll
  for (int m = 1; m < 16; m <<= 1) v += __shfl_xor(v, m, 32);
  return v;
}

// ---------------- Stage 0a: weights f32 -> bf16 ----------------
__global__ void cvt_bf16_kernel(const float* __restrict__ src,
                                unsigned short* __restrict__ dst_, long long n){
  __bf16* dst = (__bf16*)dst_;
  long long i = (long long)blockIdx.x * blockDim.x + threadIdx.x;
  long long stride = (long long)gridDim.x * blockDim.x;
  for (; i < n; i += stride) dst[i] = (__bf16)src[i];
}

// ---------------- Stage 0b: x [B,C,N] f32 -> xT [B,N,C] bf16 ----------------
__global__ __launch_bounds__(256) void cvtT_kernel(const float* __restrict__ x,
                                                   unsigned short* __restrict__ xt_){
  __bf16* xt = (__bf16*)xt_;
  __shared__ __bf16 T[64 * 65];
  const int b  = blockIdx.z;
  const int c0 = blockIdx.y * 64;
  const int n0 = blockIdx.x * 64;
#pragma unroll
  for (int i = 0; i < 16; i++){
    int lin = threadIdx.x + 256 * i;
    int cl = lin >> 6, nl = lin & 63;
    T[cl * 65 + nl] = (__bf16)x[((size_t)b * CC + c0 + cl) * NN + n0 + nl];
  }
  __syncthreads();
#pragma unroll
  for (int i = 0; i < 16; i++){
    int lin = threadIdx.x + 256 * i;
    int nl = lin >> 6, cl = lin & 63;
    xt[((size_t)b * NN + n0 + nl) * CC + c0 + cl] = T[cl * 65 + nl];
  }
}

// ---------------- Stage 1: fused QKV + anchor projections ----------------
// grid: (N/64, C/128, B*4); block 256 (8 waves).
// p=0: q -> [b][n][c] ; p=1: k -> [b][n][c] ; p=2: v -> [b][c][n] ; p=3: out = x + anchor.
__global__ __launch_bounds__(256) void proj_kernel(
    const unsigned short* __restrict__ xt_, const unsigned short* __restrict__ wbf_,
    const float* __restrict__ bq, const float* __restrict__ bk,
    const float* __restrict__ bv, const float* __restrict__ ba,
    const float* __restrict__ x,
    unsigned short* __restrict__ q_, unsigned short* __restrict__ k_,
    unsigned short* __restrict__ v_, float* __restrict__ out)
{
  const __bf16* xt  = (const __bf16*)xt_;
  const __bf16* wbf = (const __bf16*)wbf_;
  __bf16* q = (__bf16*)q_;
  __bf16* k = (__bf16*)k_;
  __bf16* v = (__bf16*)v_;

  __shared__ __align__(16) __bf16 Xs[2][64 * 32];   // double-buffered [64n][32c]

  const int b    = blockIdx.z >> 2;
  const int p    = blockIdx.z & 3;
  const int n0   = blockIdx.x * 64;
  const int wave = threadIdx.x >> 5;
  const int lane = threadIdx.x & 31;
  const int ln   = lane & 15;
  const int h16  = lane >> 4;
  const int o0   = blockIdx.y * 128 + wave * 16;

  const __bf16* W    = wbf + (size_t)p * CC * CC;
  const float*  bias = (p == 0) ? bq : (p == 1) ? bk : (p == 2) ? bv : ba;

  v8f acc[4];
#pragma unroll
  for (int r = 0; r < 8; r++){
    float bb = bias[o0 + r + 8 * h16];
#pragma unroll
    for (int ns = 0; ns < 4; ns++) acc[ns][r] = bb;
  }

  const int koff = h16 * 8;
  const int kb   = h16 * 16;
  const int orow = o0 + ln;

  // Async-issue one x chunk [64n][32c]: one b128 per thread.
  auto issueX = [&](int cs){
    int t   = threadIdx.x;
    int nl  = t >> 2;
    int cof = (t & 3) * 8;
    async_cp_b128(&Xs[cs & 1][t * 8],
                  xt + ((size_t)b * NN + n0 + nl) * CC + cs * 32 + cof);
  };

  issueX(0);
  for (int cs = 0; cs < 8; cs++){
    wait_async0();        // this wave's stores into Xs[cs&1] complete
    __syncthreads();      // all waves' stores visible; also postdates all reads
                          // of Xs[cs&1] from iteration cs-2 -> safe to refill later
    if (cs < 7) issueX(cs + 1);   // fills the *other* buffer

    ABu a;
    const v4u* wp = (const v4u*)(W + (size_t)orow * CC + cs * 32 + koff);
    a.q[0] = wp[0];
    a.q[1] = wp[2];

    const __bf16* Xb = &Xs[cs & 1][0];
#pragma unroll
    for (int ns = 0; ns < 4; ns++){
      ABu bm;
      const v4u* bp = (const v4u*)(Xb + (ns * 16 + ln) * 32 + koff);
      bm.q[0] = bp[0];
      bm.q[1] = bp[2];
      acc[ns] = __builtin_amdgcn_wmma_f32_16x16x32_bf16(
          false, a.v, false, bm.v, (short)0, acc[ns], false, false);
    }
  }

#pragma unroll
  for (int ns = 0; ns < 4; ns++){
#pragma unroll
    for (int r = 0; r < 8; r++){
      int o = o0 + r + 8 * h16;
      int n = n0 + ns * 16 + ln;
      float f = acc[ns][r];
      if (p == 0)      q[((size_t)b * NN + n) * CC + o] = (__bf16)f;
      else if (p == 1) k[((size_t)b * NN + n) * CC + o] = (__bf16)f;
      else if (p == 2) v[((size_t)b * CC + o) * NN + n] = (__bf16)f;
      else { size_t idx = ((size_t)b * CC + o) * NN + n; out[idx] = x[idx] + f; }
    }
  }
}

// ---------------- Stage 2: flash attention + residual add ----------------
// grid: (N/64, B); block 128 (4 waves); wave owns 16 query rows.
__global__ __launch_bounds__(128) void attn_kernel(
    const unsigned short* __restrict__ q_, const unsigned short* __restrict__ k_,
    const unsigned short* __restrict__ v_, float* __restrict__ out)
{
  const __bf16* q = (const __bf16*)q_;
  const __bf16* k = (const __bf16*)k_;   // [b][n][c]
  const __bf16* v = (const __bf16*)v_;   // [b][c][n]

  __shared__ __align__(16) __bf16 Vs[256 * 64];     // 32 KB [256c][64m]
  __shared__ __align__(16) __bf16 Ks[2][64 * 32];   //  8 KB double-buffered [64m][32c]
  __shared__ __align__(16) __bf16 Ps[4][16 * 64];   //  8 KB per-wave P tile

  const int b    = blockIdx.y;
  const int n0   = blockIdx.x * 64;
  const int wave = threadIdx.x >> 5;
  const int lane = threadIdx.x & 31;
  const int ln   = lane & 15;
  const int h16  = lane >> 4;
  const int koff = h16 * 8;
  const int kb   = h16 * 16;
  const int n0w  = n0 + wave * 16;

  // Preload Q A-fragments for all 8 c-steps.
  ABu Aq[8];
  {
    const __bf16* qrow = q + ((size_t)b * NN + n0w + ln) * CC;
#pragma unroll
    for (int cs = 0; cs < 8; cs++){
      const v4u* qp = (const v4u*)(qrow + cs * 32 + koff);
      Aq[cs].q[0] = qp[0];
      Aq[cs].q[1] = qp[2];
    }
  }

  v8f   Oacc[16];
  float Mi[8], li[8];
#pragma unroll
  for (int r = 0; r < 8; r++){ Mi[r] = -3.0e38f; li[r] = 0.f; }
#pragma unroll
  for (int ct = 0; ct < 16; ct++)
#pragma unroll
    for (int r = 0; r < 8; r++) Oacc[ct][r] = 0.f;

  for (int j = 0; j < 64; j++){
    const int m0 = j * 64;

    auto issueK = [&](int cs){
#pragma unroll
      for (int i = 0; i < 2; i++){
        int cc  = threadIdx.x + i * 128;
        int ml  = cc >> 2;
        int cof = (cc & 3) * 8;
        async_cp_b128(&Ks[cs & 1][cc * 8],
                      k + ((size_t)b * NN + m0 + ml) * CC + cs * 32 + cof);
      }
    };

    // Issue K chunk 0, then the whole V block [256c][64m] (16 b128/thread).
    // V stores land during the score loop; waits there cover them too.
    issueK(0);
#pragma unroll
    for (int i = 0; i < 16; i++){
      int cc  = threadIdx.x + i * 128;
      int cl  = cc >> 3;
      int mof = (cc & 7) * 8;
      async_cp_b128(&Vs[cc * 8], v + ((size_t)b * CC + cl) * NN + m0 + mof);
    }
    if (j + 1 < 64)
      __builtin_prefetch(v + ((size_t)b * CC + threadIdx.x * 2) * NN + m0 + 64, 0, 1);

    // ---- scores: S[16n x 64m] = Q . K ----
    v8f Sacc[4];
#pragma unroll
    for (int ms = 0; ms < 4; ms++)
#pragma unroll
      for (int r = 0; r < 8; r++) Sacc[ms][r] = 0.f;

    for (int cs = 0; cs < 8; cs++){
      wait_async0();
      __syncthreads();                 // publishes Ks[cs&1] (and Vs on cs==0)
      if (cs < 7) issueK(cs + 1);      // refill the other buffer

      const __bf16* Kb = &Ks[cs & 1][0];
#pragma unroll
      for (int ms = 0; ms < 4; ms++){
        ABu bm;
        const v4u* bp = (const v4u*)(Kb + (ms * 16 + ln) * 32 + koff);
        bm.q[0] = bp[0];
        bm.q[1] = bp[2];
        Sacc[ms] = __builtin_amdgcn_wmma_f32_16x16x32_bf16(
            false, Aq[cs].v, false, bm.v, (short)0, Sacc[ms], false, false);
      }
    }

    // ---- online softmax (rows live in 16-lane halves) ----
#pragma unroll
    for (int r = 0; r < 8; r++){
      float rm = fmaxf(fmaxf(Sacc[0][r], Sacc[1][r]), fmaxf(Sacc[2][r], Sacc[3][r]));
      rm = half_max16(rm);
      float Mn    = fmaxf(Mi[r], rm);
      float alpha = __expf(Mi[r] - Mn);
      float ps = 0.f;
#pragma unroll
      for (int ms = 0; ms < 4; ms++){
        float pv = __expf(Sacc[ms][r] - Mn);
        Sacc[ms][r] = pv;
        ps += pv;
      }
      ps = half_sum16(ps);
      li[r] = li[r] * alpha + ps;
      Mi[r] = Mn;
#pragma unroll
      for (int ct = 0; ct < 16; ct++) Oacc[ct][r] *= alpha;
    }

    // ---- C-layout -> A-layout for P via per-wave LDS round trip ----
    __bf16* Pw = &Ps[wave][0];
#pragma unroll
    for (int ms = 0; ms < 4; ms++)
#pragma unroll
      for (int r = 0; r < 8; r++)
        Pw[(r + 8 * h16) * 64 + ms * 16 + ln] = (__bf16)Sacc[ms][r];
    // Same-wave LDS ops are in-order: no barrier needed for the read-back.

    ABu Ap[2];
#pragma unroll
    for (int msr = 0; msr < 2; msr++){
      const v4u* pp = (const v4u*)(Pw + (size_t)ln * 64 + msr * 32 + koff);
      Ap[msr].q[0] = pp[0];
      Ap[msr].q[1] = pp[2];
    }

    // ---- O[16n x 256c] += P . V^T ----
#pragma unroll
    for (int ct = 0; ct < 16; ct++){
#pragma unroll
      for (int msr = 0; msr < 2; msr++){
        ABu bm;
        const v4u* bp = (const v4u*)(Vs + (ct * 16 + ln) * 64 + msr * 32 + koff);
        bm.q[0] = bp[0];
        bm.q[1] = bp[2];
        Oacc[ct] = __builtin_amdgcn_wmma_f32_16x16x32_bf16(
            false, Ap[msr].v, false, bm.v, (short)0, Oacc[ct], false, false);
      }
    }
    __syncthreads();   // all waves done reading Vs/Ks before next block refills
  }

  // Finalize: divide by running sum, add into out (= x + anchor from stage 1).
#pragma unroll
  for (int r = 0; r < 8; r++){
    float inv = 1.0f / li[r];
    int n = n0w + r + 8 * h16;
#pragma unroll
    for (int ct = 0; ct < 16; ct++){
      int c = ct * 16 + ln;
      size_t idx = ((size_t)b * CC + c) * NN + n;
      out[idx] += Oacc[ct][r] * inv;
    }
  }
}

// ---------------- host launcher ----------------
extern "C" void kernel_launch(void* const* d_in, const int* in_sizes, int n_in,
                              void* d_out, int out_size, void* d_ws, size_t ws_size,
                              hipStream_t stream)
{
  const float* x  = (const float*)d_in[0];
  const float* Wq = (const float*)d_in[1];
  const float* bq = (const float*)d_in[2];
  const float* Wk = (const float*)d_in[3];
  const float* bk = (const float*)d_in[4];
  const float* Wv = (const float*)d_in[5];
  const float* bv = (const float*)d_in[6];
  const float* Wa = (const float*)d_in[7];
  const float* ba = (const float*)d_in[8];
  float* out = (float*)d_out;

  char* ws = (char*)d_ws;
  size_t off = 0;
  unsigned short* xt  = (unsigned short*)(ws + off); off += (size_t)BB * NN * CC * 2;
  unsigned short* wbf = (unsigned short*)(ws + off); off += (size_t)4 * CC * CC * 2;
  unsigned short* qb  = (unsigned short*)(ws + off); off += (size_t)BB * NN * CC * 2;
  unsigned short* kbf = (unsigned short*)(ws + off); off += (size_t)BB * NN * CC * 2;
  unsigned short* vb  = (unsigned short*)(ws + off); off += (size_t)BB * CC * NN * 2;
  (void)off; (void)ws_size; (void)in_sizes; (void)n_in; (void)out_size;

  cvt_bf16_kernel<<<64, 256, 0, stream>>>(Wq, wbf + 0 * CC * CC, CC * CC);
  cvt_bf16_kernel<<<64, 256, 0, stream>>>(Wk, wbf + 1 * CC * CC, CC * CC);
  cvt_bf16_kernel<<<64, 256, 0, stream>>>(Wv, wbf + 2 * CC * CC, CC * CC);
  cvt_bf16_kernel<<<64, 256, 0, stream>>>(Wa, wbf + 3 * CC * CC, CC * CC);

  dim3 g0(NN / 64, CC / 64, BB);
  cvtT_kernel<<<g0, 256, 0, stream>>>(x, xt);

  dim3 g1(NN / 64, CC / 128, BB * 4);
  proj_kernel<<<g1, 256, 0, stream>>>(xt, wbf, bq, bk, bv, ba, x, qb, kbf, vb, out);

  dim3 g2(NN / 64, BB);
  attn_kernel<<<g2, 128, 0, stream>>>(qb, kbf, vb, out);
}
